// CrossAttention_17085379903982
// MI455X (gfx1250) — matrix operand order repaired and checked by reading, not compiled, not run on hardware
//
#include <hip/hip_runtime.h>
#include <hip/hip_bf16.h>

// ---------- types ----------
typedef __bf16 v16bf __attribute__((ext_vector_type(16)));
typedef __bf16 v8bf  __attribute__((ext_vector_type(8)));
typedef float  v8f   __attribute__((ext_vector_type(8)));

union AB16 { v16bf v; v8bf h[2]; };

static __device__ __forceinline__ v8f wmma_bf16(v16bf a, v16bf b, v8f c) {
  return __builtin_amdgcn_wmma_f32_16x16x32_bf16(false, a, false, b, (short)0, c, false, false);
}

// ---------- constants ----------
#define HN   8
#define DH   64
#define NQ   2048
#define NKV  2048

// ---------- fp32 -> bf16 convert ----------
__global__ void cvt_kernel(const float* __restrict__ src, __bf16* __restrict__ dst, long n) {
  long i = (long)blockIdx.x * blockDim.x + threadIdx.x;
  if (i < n) dst[i] = (__bf16)src[i];
}

// fp32 (R x C) -> bf16 transposed (C x R)
__global__ void tcvt_kernel(const float* __restrict__ src, __bf16* __restrict__ dst, int R, int C) {
  long i = (long)blockIdx.x * blockDim.x + threadIdx.x;
  if (i < (long)R * C) {
    int r = (int)(i / C);
    int c = (int)(i % C);
    dst[(long)c * R + r] = (__bf16)src[i];
  }
}

// ---------- generic bf16 WMMA GEMM ----------
// C(MxN) = A(MxK) * B(KxN), B given transposed (BT: N x K, row-major).
// Block = 8 waves; block tile 256x64; wave tile 32x64 (2 M-frags x 4 N-tiles,
// 8 WMMAs per k-step, 1.5 VMEM ops per WMMA).
// Output indexing: m -> (b = m/rowsPerBatch, i = m%rowsPerBatch),
//                  n -> (h = n/64, d = n%64); off = b*sB + i*sM + h*sH + d*sD.
__global__ __launch_bounds__(256) void gemm_bf16_kernel(
    const __bf16* __restrict__ A, const __bf16* __restrict__ BT,
    int M, int N, int K,
    __bf16* __restrict__ outb, float* __restrict__ outf,
    const float* __restrict__ bias,
    long sB, long sM, long sH, long sD, int rowsPerBatch, float scale)
{
  const int lane = threadIdx.x & 31;
  const int w    = threadIdx.x >> 5;
  const int half = lane >> 4;
  const int nl   = lane & 15;

  const int m0 = blockIdx.x * 256 + w * 32;
  const int n0 = blockIdx.y * 64;

  v8f acc[2][4] = {};
  const __bf16* arow0 = A + (long)(m0 + nl) * K;
  const __bf16* arow1 = A + (long)(m0 + 16 + nl) * K;

  for (int k0 = 0; k0 < K; k0 += 32) {
    AB16 a0, a1;
    a0.h[0] = *(const v8bf*)(arow0 + k0 + half * 16);       // K = k0+half*16 .. +7
    a0.h[1] = *(const v8bf*)(arow0 + k0 + 8 + half * 16);   // K = k0+8+half*16 .. +7
    a1.h[0] = *(const v8bf*)(arow1 + k0 + half * 16);
    a1.h[1] = *(const v8bf*)(arow1 + k0 + 8 + half * 16);
#pragma unroll
    for (int t = 0; t < 4; ++t) {
      const __bf16* brow = BT + (long)(n0 + t * 16 + nl) * K + k0 + half * 16;
      v16bf b = *(const v16bf*)brow;                         // K = k0+half*16 .. +15
      acc[0][t] = wmma_bf16(a0.v, b, acc[0][t]);
      acc[1][t] = wmma_bf16(a1.v, b, acc[1][t]);
    }
  }

#pragma unroll
  for (int mt = 0; mt < 2; ++mt) {
#pragma unroll
    for (int t = 0; t < 4; ++t) {
#pragma unroll
      for (int r = 0; r < 8; ++r) {
        int mg = m0 + mt * 16 + r + 8 * half;  // C layout: row M = vgpr + 8*lanehalf
        int ng = n0 + t * 16 + nl;             //          col N = lane % 16
        long off = (long)(mg / rowsPerBatch) * sB + (long)(mg % rowsPerBatch) * sM
                 + (long)(ng >> 6) * sH + (long)(ng & 63) * sD;
        float v = acc[mt][t][r] * scale;
        if (outf) outf[off] = v + (bias ? bias[ng] : 0.0f);
        else      outb[off] = (__bf16)v;
      }
    }
  }
}

// ---------- flash attention ----------
// Q: (B,H,Nq,D) bf16 (pre-scaled by 1/sqrt(D));  K: (B,H,Nkv,D) bf16;
// VT: (B,H,D,Nkv) bf16;  O: (B*Nq, H*D) bf16.
// grid = (B*H, Nq/128), block = 256 (8 waves); each wave: 16 query rows.
__global__ __launch_bounds__(256) void attn_kernel(
    const __bf16* __restrict__ Qb, const __bf16* __restrict__ Kb,
    const __bf16* __restrict__ VTb, __bf16* __restrict__ Ob)
{
  __shared__ __bf16 pbuf[8][16 * 32];   // per-wave P staging (C-layout -> A-frag reshape)

  const int lane = threadIdx.x & 31;
  const int w    = threadIdx.x >> 5;
  const int half = lane >> 4;
  const int nl   = lane & 15;

  const int bh = blockIdx.x;
  const int b  = bh >> 3;
  const int h  = bh & 7;
  const int i0 = blockIdx.y * 128 + w * 16;

  const __bf16* Qh = Qb  + (long)bh * NQ  * DH;
  const __bf16* Kh = Kb  + (long)bh * NKV * DH;
  const __bf16* Vh = VTb + (long)bh * DH  * NKV;

  // Q fragments for the two K-steps over D=64 (held in registers for whole loop)
  AB16 aq[2];
  const __bf16* qrow = Qh + (long)(i0 + nl) * DH;
#pragma unroll
  for (int ks = 0; ks < 2; ++ks) {
    aq[ks].h[0] = *(const v8bf*)(qrow + ks * 32 + half * 16);
    aq[ks].h[1] = *(const v8bf*)(qrow + ks * 32 + 8 + half * 16);
  }

  v8f acc[4] = {};
  float mrow[8], lrow[8];
#pragma unroll
  for (int r = 0; r < 8; ++r) { mrow[r] = -1e30f; lrow[r] = 0.0f; }

  __bf16* pw = &pbuf[w][0];

  for (int kv0 = 0; kv0 < NKV; kv0 += 32) {
    // ---- S = Q * K^T for 16x32 score strip (two 16x16 N-tiles) ----
    v8f s0 = {}, s1 = {};
#pragma unroll
    for (int ks = 0; ks < 2; ++ks) {
      v16bf b0 = *(const v16bf*)(Kh + (long)(kv0 + nl)      * DH + ks * 32 + half * 16);
      v16bf b1 = *(const v16bf*)(Kh + (long)(kv0 + 16 + nl) * DH + ks * 32 + half * 16);
      s0 = wmma_bf16(aq[ks].v, b0, s0);
      s1 = wmma_bf16(aq[ks].v, b1, s1);
    }

    // ---- online softmax (fp32); row M = r + 8*half, cols across 16 lanes ----
#pragma unroll
    for (int r = 0; r < 8; ++r) {
      float x0 = s0[r], x1 = s1[r];
      float mx = fmaxf(x0, x1);
#pragma unroll
      for (int off = 1; off < 16; off <<= 1) mx = fmaxf(mx, __shfl_xor(mx, off, 32));
      float mnew  = fmaxf(mrow[r], mx);
      float alpha = __expf(mrow[r] - mnew);
      float p0 = __expf(x0 - mnew);
      float p1 = __expf(x1 - mnew);
      float ps = p0 + p1;
#pragma unroll
      for (int off = 1; off < 16; off <<= 1) ps += __shfl_xor(ps, off, 32);
      lrow[r] = lrow[r] * alpha + ps;
      mrow[r] = mnew;
#pragma unroll
      for (int t = 0; t < 4; ++t) acc[t][r] *= alpha;
      int m = r + 8 * half;
      pw[m * 32 + nl]      = (__bf16)p0;
      pw[m * 32 + 16 + nl] = (__bf16)p1;
    }

    // ---- reshape P (row-major in LDS) into A-fragment; DS ops are in-order per wave ----
    AB16 ap;
    ap.h[0] = *(const v8bf*)(pw + nl * 32 + half * 16);
    ap.h[1] = *(const v8bf*)(pw + nl * 32 + 8 + half * 16);

    // ---- O += P * V  (V^T rows are contiguous in kv) ----
#pragma unroll
    for (int t = 0; t < 4; ++t) {
      v16bf bv = *(const v16bf*)(Vh + (long)(t * 16 + nl) * NKV + kv0 + half * 16);
      acc[t] = wmma_bf16(ap.v, bv, acc[t]);
    }
  }

  // ---- normalize and store O (B*Nq, H*D) ----
  const long obase = (long)b * NQ * (HN * DH) + (long)h * DH;
#pragma unroll
  for (int t = 0; t < 4; ++t) {
#pragma unroll
    for (int r = 0; r < 8; ++r) {
      int m = r + 8 * half;
      float v = acc[t][r] / lrow[r];
      Ob[obase + (long)(i0 + m) * (HN * DH) + t * 16 + nl] = (__bf16)v;
    }
  }
}

// ---------- launcher ----------
extern "C" void kernel_launch(void* const* d_in, const int* in_sizes, int n_in,
                              void* d_out, int out_size, void* d_ws, size_t ws_size,
                              hipStream_t stream) {
  (void)in_sizes; (void)n_in; (void)out_size; (void)ws_size;

  const float* x   = (const float*)d_in[0];   // (2,2048,1024)
  const float* ctx = (const float*)d_in[1];   // (2,2048,768)
  const float* Wq  = (const float*)d_in[2];   // (1024,512)
  const float* Wk  = (const float*)d_in[3];   // (768,512)
  const float* Wv  = (const float*)d_in[4];   // (768,512)
  const float* Wo  = (const float*)d_in[5];   // (512,1024)
  const float* bo  = (const float*)d_in[6];   // (1024,)
  float* out = (float*)d_out;                 // (2,2048,1024)

  const long nX  = 2L * NQ * 1024;   // 4194304
  const long nC  = 2L * NKV * 768;   // 3145728
  const long nWq = 1024L * 512;
  const long nWk = 768L * 512;
  const long nWo = 512L * 1024;
  const long nQ  = 2L * HN * NQ * DH;   // 2097152
  const long nO  = 4096L * 512;

  char* ws = (char*)d_ws;
  size_t off = 0;
  auto carve = [&](size_t bytes) { char* p = ws + off; off += (bytes + 255) & ~(size_t)255; return p; };
  __bf16* xb   = (__bf16*)carve(nX  * 2);
  __bf16* cb   = (__bf16*)carve(nC  * 2);
  __bf16* WqT  = (__bf16*)carve(nWq * 2);
  __bf16* WkT  = (__bf16*)carve(nWk * 2);
  __bf16* WvT  = (__bf16*)carve(nWk * 2);
  __bf16* WoT  = (__bf16*)carve(nWo * 2);
  __bf16* Qb   = (__bf16*)carve(nQ  * 2);
  __bf16* Kb   = (__bf16*)carve(nQ  * 2);
  __bf16* VTb  = (__bf16*)carve(nQ  * 2);
  __bf16* Ob   = (__bf16*)carve(nO  * 2);

  // 1) convert activations; transpose+convert weights
  cvt_kernel<<<(int)((nX + 255) / 256), 256, 0, stream>>>(x,   xb, nX);
  cvt_kernel<<<(int)((nC + 255) / 256), 256, 0, stream>>>(ctx, cb, nC);
  tcvt_kernel<<<(int)((nWq + 255) / 256), 256, 0, stream>>>(Wq, WqT, 1024, 512);
  tcvt_kernel<<<(int)((nWk + 255) / 256), 256, 0, stream>>>(Wk, WkT, 768, 512);
  tcvt_kernel<<<(int)((nWk + 255) / 256), 256, 0, stream>>>(Wv, WvT, 768, 512);
  tcvt_kernel<<<(int)((nWo + 255) / 256), 256, 0, stream>>>(Wo, WoT, 512, 1024);

  const float SCALE = 0.125f;  // 64^-0.5
  const long sBqk = (long)HN * NQ * DH;  // 1048576
  const long sHqk = (long)NQ * DH;       // 131072

  // 2) Q = x@Wq (scaled), layout (B,H,Nq,D)
  gemm_bf16_kernel<<<dim3(16, 8), 256, 0, stream>>>(
      xb, WqT, 4096, 512, 1024, Qb, nullptr, nullptr,
      sBqk, /*sM=*/DH, sHqk, /*sD=*/1, NQ, SCALE);
  // 3) K = ctx@Wk, layout (B,H,Nkv,D)
  gemm_bf16_kernel<<<dim3(16, 8), 256, 0, stream>>>(
      cb, WkT, 4096, 512, 768, Kb, nullptr, nullptr,
      sBqk, /*sM=*/DH, sHqk, /*sD=*/1, NKV, 1.0f);
  // 4) V = ctx@Wv, stored transposed: (B,H,D,Nkv)
  gemm_bf16_kernel<<<dim3(16, 8), 256, 0, stream>>>(
      cb, WvT, 4096, 512, 768, VTb, nullptr, nullptr,
      sBqk, /*sM=*/1, sHqk, /*sD=*/NKV, NKV, 1.0f);

  // 5) flash attention -> Ob (B*Nq, 512)
  attn_kernel<<<dim3(2 * HN, NQ / 128), 256, 0, stream>>>(Qb, Kb, VTb, Ob);

  // 6) out = Ob@Wo + bo -> fp32 d_out (B*Nq, 1024)
  gemm_bf16_kernel<<<dim3(16, 16), 256, 0, stream>>>(
      Ob, WoT, 4096, 1024, 512, nullptr, out, bo,
      (long)NQ * 1024, /*sM=*/1024, /*sH=*/64, /*sD=*/1, NQ, 1.0f);
}
